// FARGANSub_17016660426803
// MI455X (gfx1250) — compile-verified, matrix operand order checked by reading, not run on hardware
//
#include <hip/hip_runtime.h>

#define BDIM 256
#define ROWS 16
#define CONDN 256
#define SUBN 40
#define MEMN 256
#define INK 376
#define INKP 384

typedef __bf16 bf16_t;
typedef bf16_t v16bf __attribute__((ext_vector_type(16)));
typedef float  v8f   __attribute__((ext_vector_type(8)));

// ---- packed-weight layout offsets (dwords) ----
// fragment-major bf16: [tile][kstep][lane 0..31][8 dwords]; one (tile,kstep) = 1KB.
#define SZ_D1  (16 * 12 * 256)   // W_d1: N=256, K=376->12 ksteps
#define SZ_SQ  (16 * 8 * 256)    // 256x256
#define SZ_IH  (48 * 8 * 256)    // 768x256
#define SZ_OUT (3 * 8 * 256)     // 40x256 (rows padded to 48)

#define P_WD1  0
#define P_GD1  (P_WD1  + SZ_D1)
#define P_WD2  (P_GD1  + SZ_SQ)
#define P_GD2  (P_WD2  + SZ_SQ)
#define P_WIH1 (P_GD2  + SZ_SQ)
#define P_WHH1 (P_WIH1 + SZ_IH)
#define P_GG1  (P_WHH1 + SZ_IH)
#define P_WIH2 (P_GG1  + SZ_SQ)
#define P_WHH2 (P_WIH2 + SZ_IH)
#define P_GG2  (P_WHH2 + SZ_IH)
#define P_WIH3 (P_GG2  + SZ_SQ)
#define P_WHH3 (P_WIH3 + SZ_IH)
#define P_GG3  (P_WHH3 + SZ_IH)
#define P_WOUT (P_GG3  + SZ_SQ)

// Branch-free fast activations (hardware exp + rcp; no EXEC-mask branches).
static __device__ __forceinline__ float fast_sigmoid(float x) {
    return __builtin_amdgcn_rcpf(1.0f + __expf(-x));
}
static __device__ __forceinline__ float fast_tanh(float x) {
    return 2.0f * __builtin_amdgcn_rcpf(1.0f + __expf(-2.0f * x)) - 1.0f;
}

union FragU  { unsigned u[8]; v16bf v; };
union FragQ  { uint4 q[2]; v16bf v; };

// A-fragment (16x32 bf16, MxK) from LDS bf16 row-major, leading dim ld (merges to 2x ds_load_b128).
static __device__ __forceinline__ v16bf load_a_lds(const bf16_t* X, int ld, int kk, int lane) {
    int m  = lane & 15;
    int kb = kk + ((lane >> 4) << 3);
    const unsigned short* Xs = (const unsigned short*)X;
    FragU fr;
#pragma unroll
    for (int i = 0; i < 8; ++i) {
        int k = kb + ((i & 3) << 1) + ((i >> 2) << 4);
        fr.u[i] = *(const unsigned*)(Xs + m * ld + k);
    }
    return fr.v;
}

// B-fragment from pre-packed bf16 weights: two coalesced b128 loads, no guards, no converts.
static __device__ __forceinline__ v16bf load_b_pk(const unsigned* P, int tile, int nks, int ks, int lane) {
    const unsigned* p = P + (((size_t)tile * nks + ks) * 32 + lane) * 8;
    FragQ fr;
    fr.q[0] = *(const uint4*)(p);
    fr.q[1] = *(const uint4*)(p + 4);
    return fr.v;
}

static __device__ __forceinline__ v8f wmma_bf(v16bf a, v16bf b, v8f c) {
    return __builtin_amdgcn_wmma_f32_16x16x32_bf16(false, a, false, b, (short)0, c, false, false);
}

#define V8F_ZERO {0.f,0.f,0.f,0.f,0.f,0.f,0.f,0.f}

// ---------------- weight prepack: fp32 [N][K] -> fragment-major bf16 ----------------
__global__ __launch_bounds__(256)
void prepack_kernel(const float* __restrict__ W, unsigned* __restrict__ P, int N, int K) {
    int nk = (K + 31) >> 5;
    int nt = (N + 15) >> 4;
    int total = nt * nk * 32;
    int tid = blockIdx.x * blockDim.x + threadIdx.x;
    if (tid >= total) return;
    int lane = tid & 31;
    int ks   = (tid >> 5) % nk;
    int tile = tid / (nk * 32);
    int r  = tile * 16 + (lane & 15);
    bool rok = (r < N);
    const float* wr = W + (size_t)(rok ? r : 0) * K;
    int kb = (ks << 5) + ((lane >> 4) << 3);
    unsigned* dst = P + (size_t)tid * 8;
#pragma unroll
    for (int i = 0; i < 8; ++i) {
        int k = kb + ((i & 3) << 1) + ((i >> 2) << 4);
        float lo = (rok && (k     < K)) ? wr[k]     : 0.0f;
        float hi = (rok && (k + 1 < K)) ? wr[k + 1] : 0.0f;
        union { bf16_t b[2]; unsigned u; } pk;
        pk.b[0] = (bf16_t)lo;
        pk.b[1] = (bf16_t)hi;
        dst[i] = pk.u;
    }
}

// ---------------- fused layer helpers ----------------

// Y[16][256](bf16 LDS) = tanh(X[16][K] @ W^T), W packed (16 tiles x nks ksteps)
static __device__ __forceinline__ void gemm_tanh(const bf16_t* X, int ldx, int nks,
                                                 const unsigned* P,
                                                 bf16_t* Y, int wv, int lane) {
    for (int s = 0; s < 2; ++s) {
        int t = wv + (s << 3);
        v8f acc = V8F_ZERO;
        for (int ks = 0; ks < nks; ++ks) {
            v16bf a = load_a_lds(X, ldx, ks << 5, lane);
            v16bf b = load_b_pk(P, t, nks, ks, lane);
            acc = wmma_bf(a, b, acc);
        }
        int m0 = (lane >> 4) << 3;
        int n  = (lane & 15) + t * 16;
#pragma unroll
        for (int v = 0; v < 8; ++v)
            Y[(m0 + v) * CONDN + n] = (bf16_t)fast_tanh(acc[v]);
    }
}

// Y = X * sigmoid(X @ G^T), all [16][256], G packed 16x8
static __device__ __forceinline__ void glu_layer(const bf16_t* X, const unsigned* P,
                                                 bf16_t* Y, int wv, int lane) {
    for (int s = 0; s < 2; ++s) {
        int t = wv + (s << 3);
        v8f acc = V8F_ZERO;
        for (int ks = 0; ks < 8; ++ks) {
            v16bf a = load_a_lds(X, CONDN, ks << 5, lane);
            v16bf b = load_b_pk(P, t, 8, ks, lane);
            acc = wmma_bf(a, b, acc);
        }
        int m0 = (lane >> 4) << 3;
        int n  = (lane & 15) + t * 16;
#pragma unroll
        for (int v = 0; v < 8; ++v) {
            float x = (float)X[(m0 + v) * CONDN + n];
            Y[(m0 + v) * CONDN + n] = (bf16_t)(x * fast_sigmoid(acc[v]));
        }
    }
}

// GRU cell: gates [r,z,n]; h' = (1-z)*n + z*h. Wih/Whh packed 48x8 (tile = gate*16 + t).
static __device__ __forceinline__ void gru_layer(const bf16_t* X, const bf16_t* H,
                                                 const unsigned* Pih, const unsigned* Phh,
                                                 bf16_t* Y, float* gout,
                                                 int row0, int wv, int lane) {
    for (int s = 0; s < 2; ++s) {
        int t = wv + (s << 3);
        v8f a_ir = V8F_ZERO, a_iz = V8F_ZERO, a_in = V8F_ZERO;
        v8f a_hr = V8F_ZERO, a_hz = V8F_ZERO, a_hn = V8F_ZERO;
        for (int ks = 0; ks < 8; ++ks) {
            v16bf ax = load_a_lds(X, CONDN, ks << 5, lane);
            v16bf ah = load_a_lds(H, CONDN, ks << 5, lane);
            a_ir = wmma_bf(ax, load_b_pk(Pih,      t, 8, ks, lane), a_ir);
            a_iz = wmma_bf(ax, load_b_pk(Pih, 16 + t, 8, ks, lane), a_iz);
            a_in = wmma_bf(ax, load_b_pk(Pih, 32 + t, 8, ks, lane), a_in);
            a_hr = wmma_bf(ah, load_b_pk(Phh,      t, 8, ks, lane), a_hr);
            a_hz = wmma_bf(ah, load_b_pk(Phh, 16 + t, 8, ks, lane), a_hz);
            a_hn = wmma_bf(ah, load_b_pk(Phh, 32 + t, 8, ks, lane), a_hn);
        }
        int m0 = (lane >> 4) << 3;
        int n  = (lane & 15) + t * 16;
#pragma unroll
        for (int v = 0; v < 8; ++v) {
            int m = m0 + v;
            float r  = fast_sigmoid(a_ir[v] + a_hr[v]);
            float z  = fast_sigmoid(a_iz[v] + a_hz[v]);
            float nn = fast_tanh(a_in[v] + r * a_hn[v]);
            float ho = (float)H[m * CONDN + n];
            float hn = (1.0f - z) * nn + z * ho;
            Y[m * CONDN + n] = (bf16_t)hn;
            gout[(size_t)(row0 + m) * CONDN + n] = hn;
        }
    }
}

// ---------------- main fused kernel ----------------
__global__ __launch_bounds__(BDIM)
void fargan_sub_kernel(const float* __restrict__ cond,
                       const float* __restrict__ exc_mem,
                       const float* __restrict__ phase,
                       const int*   __restrict__ period,
                       const float* __restrict__ gru1_state,
                       const float* __restrict__ gru2_state,
                       const float* __restrict__ gru3_state,
                       const unsigned* __restrict__ pw,   // packed weights in d_ws
                       float* __restrict__ out, int Bn)
{
    __shared__ __align__(16) bf16_t s_in[ROWS * INKP];   // 12 KB
    __shared__ __align__(16) bf16_t s_x [ROWS * CONDN];  // 8 KB
    __shared__ __align__(16) bf16_t s_y [ROWS * CONDN];  // 8 KB
    __shared__ __align__(16) bf16_t s_h [ROWS * CONDN];  // 8 KB

    const int tid  = threadIdx.x;
    const int lane = tid & 31;
    const int wv   = tid >> 5;
    const int row0 = blockIdx.x * ROWS;

    float* sig_out = out;
    float* exc_out = out + (size_t)Bn * SUBN;
    float* g1_out  = out + (size_t)Bn * (SUBN + MEMN);
    float* g2_out  = g1_out + (size_t)Bn * CONDN;
    float* g3_out  = g2_out + (size_t)Bn * CONDN;

    // ---- stage 0: gather + concat input (bf16), exc_mem shift copy, load h1 ----
    for (int i = tid; i < ROWS * INKP; i += BDIM) {
        int r = i / INKP, c = i - r * INKP;
        int gr = row0 + r;
        float val;
        if (c < CONDN) {
            val = cond[(size_t)gr * CONDN + c];
        } else if (c < CONDN + SUBN) {
            int j = c - CONDN;
            int p = period[gr];
            int lag = p > SUBN ? p : SUBN;
            val = exc_mem[(size_t)gr * MEMN + (MEMN - lag) + j];
        } else if (c < INK) {
            val = phase[(size_t)gr * (2 * SUBN) + (c - CONDN - SUBN)];
        } else {
            val = 0.0f;  // pad 376..383
        }
        s_in[i] = (bf16_t)val;
    }
    for (int i = tid; i < ROWS * (MEMN - SUBN); i += BDIM) {
        int r = i / (MEMN - SUBN), c = i - r * (MEMN - SUBN);
        exc_out[(size_t)(row0 + r) * MEMN + c] =
            exc_mem[(size_t)(row0 + r) * MEMN + SUBN + c];
    }
    for (int i = tid; i < ROWS * CONDN; i += BDIM) {
        int r = i >> 8, c = i & 255;
        s_h[i] = (bf16_t)gru1_state[(size_t)(row0 + r) * CONDN + c];
    }
    __syncthreads();

    // ---- feature dense 1 + GLU ----
    gemm_tanh(s_in, INKP, 12, pw + P_WD1, s_x, wv, lane);
    __syncthreads();
    glu_layer(s_x, pw + P_GD1, s_y, wv, lane);
    __syncthreads();

    // ---- feature dense 2 + GLU ----
    gemm_tanh(s_y, CONDN, 8, pw + P_WD2, s_x, wv, lane);
    __syncthreads();
    glu_layer(s_x, pw + P_GD2, s_y, wv, lane);
    __syncthreads();

    // ---- GRU1 ----
    gru_layer(s_y, s_h, pw + P_WIH1, pw + P_WHH1, s_x, g1_out, row0, wv, lane);
    __syncthreads();
    for (int i = tid; i < ROWS * CONDN; i += BDIM) {          // preload h2
        int r = i >> 8, c = i & 255;
        s_h[i] = (bf16_t)gru2_state[(size_t)(row0 + r) * CONDN + c];
    }
    glu_layer(s_x, pw + P_GG1, s_y, wv, lane);
    __syncthreads();

    // ---- GRU2 ----
    gru_layer(s_y, s_h, pw + P_WIH2, pw + P_WHH2, s_x, g2_out, row0, wv, lane);
    __syncthreads();
    for (int i = tid; i < ROWS * CONDN; i += BDIM) {          // preload h3
        int r = i >> 8, c = i & 255;
        s_h[i] = (bf16_t)gru3_state[(size_t)(row0 + r) * CONDN + c];
    }
    glu_layer(s_x, pw + P_GG2, s_y, wv, lane);
    __syncthreads();

    // ---- GRU3 ----
    gru_layer(s_y, s_h, pw + P_WIH3, pw + P_WHH3, s_x, g3_out, row0, wv, lane);
    __syncthreads();
    glu_layer(s_x, pw + P_GG3, s_y, wv, lane);
    __syncthreads();

    // ---- output head: sig_out[16][40] = tanh(s_y @ W_out^T), tiles on waves 0..2 ----
    if (wv < 3) {
        int t = wv;
        v8f acc = V8F_ZERO;
        for (int ks = 0; ks < 8; ++ks) {
            v16bf a = load_a_lds(s_y, CONDN, ks << 5, lane);
            v16bf b = load_b_pk(pw + P_WOUT, t, 8, ks, lane);
            acc = wmma_bf(a, b, acc);
        }
        int m0 = (lane >> 4) << 3;
        int n  = (lane & 15) + t * 16;
        if (n < SUBN) {
#pragma unroll
            for (int v = 0; v < 8; ++v) {
                float sg = fast_tanh(acc[v]);
                int gr = row0 + m0 + v;
                sig_out[(size_t)gr * SUBN + n] = sg;
                exc_out[(size_t)gr * MEMN + (MEMN - SUBN) + n] = sg;
            }
        }
    }
}

extern "C" void kernel_launch(void* const* d_in, const int* in_sizes, int n_in,
                              void* d_out, int out_size, void* d_ws, size_t ws_size,
                              hipStream_t stream) {
    (void)n_in; (void)out_size; (void)ws_size;
    const int Bn = in_sizes[0] / CONDN;       // batch size (16384)
    unsigned* pw = (unsigned*)d_ws;

    // ---- prepack all weights into fragment-major bf16 (runs every call; deterministic) ----
    struct PP { int idx; size_t off; int N, K; };
    const PP pp[14] = {
        {  9, P_WD1,  CONDN, INK   },   // W_d1  [256][376]
        { 10, P_GD1,  CONDN, CONDN },   // G_d1
        { 11, P_WD2,  CONDN, CONDN },   // W_d2
        { 12, P_GD2,  CONDN, CONDN },   // G_d2
        { 13, P_WIH1, 3*CONDN, CONDN }, // Wih1
        { 14, P_WHH1, 3*CONDN, CONDN }, // Whh1
        { 15, P_GG1,  CONDN, CONDN },   // G_g1
        { 16, P_WIH2, 3*CONDN, CONDN },
        { 17, P_WHH2, 3*CONDN, CONDN },
        { 18, P_GG2,  CONDN, CONDN },
        { 19, P_WIH3, 3*CONDN, CONDN },
        { 20, P_WHH3, 3*CONDN, CONDN },
        { 21, P_GG3,  CONDN, CONDN },
        { 22, P_WOUT, SUBN,  CONDN },   // W_out [40][256]
    };
    for (int i = 0; i < 14; ++i) {
        int nt = (pp[i].N + 15) / 16, nk = (pp[i].K + 31) / 32;
        int total = nt * nk * 32;
        prepack_kernel<<<(total + 255) / 256, 256, 0, stream>>>(
            (const float*)d_in[pp[i].idx], pw + pp[i].off, pp[i].N, pp[i].K);
    }

    // ---- fused forward ----
    dim3 grid(Bn / ROWS), block(BDIM);
    fargan_sub_kernel<<<grid, block, 0, stream>>>(
        (const float*)d_in[0],   // cond
        (const float*)d_in[2],   // exc_mem
        (const float*)d_in[3],   // phase
        (const int*)  d_in[4],   // period (int32)
        (const float*)d_in[5],   // gru1_state
        (const float*)d_in[6],   // gru2_state
        (const float*)d_in[7],   // gru3_state
        pw,
        (float*)d_out, Bn);
}
